// Attention_14139032338925
// MI455X (gfx1250) — matrix operand order repaired and checked by reading, not compiled, not run on hardware
//
#include <hip/hip_runtime.h>
#include <hip/hip_bf16.h>

typedef __attribute__((ext_vector_type(16))) _Float16 v16h;
typedef __attribute__((ext_vector_type(8)))  _Float16 v8h;
typedef __attribute__((ext_vector_type(4)))  _Float16 v4h;
typedef __attribute__((ext_vector_type(8)))  float    v8f;
typedef __attribute__((ext_vector_type(4)))  float    v4f;
typedef __attribute__((ext_vector_type(4)))  int      v4i;

#define S_LEN 4096
#define D_KEY 64
#define NQ    16      // query rows per workgroup
#define KT    256     // 16-wide key tiles
#define WAVES 8
#define BH    16      // B*H

// LDS: [16][4096] f16 scores/exp (128KB) | [16][64] f32 acc (4KB) | [16] f32 inv_sum
#define SH_BYTES (NQ*S_LEN*2 + NQ*D_KEY*4 + NQ*4)

// ---------------- pre-pass: K -> f16 (row major), V -> f16 transposed [bh][d][k] ----------------
__global__ __launch_bounds__(256, 1)
void cvt_kernel(const float* __restrict__ K,
                const float* __restrict__ V,
                _Float16* __restrict__ K16,
                _Float16* __restrict__ VT)
{
    __shared__ _Float16 tile[64][72];                  // padded to dodge bank conflicts
    const int tid = threadIdx.x;
    const int bh  = blockIdx.x >> 6;                   // 0..15
    const int kt  = blockIdx.x & 63;                   // 64-row tile index
    const size_t base = ((size_t)bh * S_LEN + kt * 64) * D_KEY;

    #pragma unroll
    for (int it = 0; it < 16; ++it) {
        const int e = tid + 256 * it;                  // coalesced f32 reads
        const int k = e >> 6, d = e & 63;
        K16[base + e] = (_Float16)K[base + e];
        tile[k][d]    = (_Float16)V[base + e];
    }
    __syncthreads();
    #pragma unroll
    for (int it = 0; it < 16; ++it) {
        const int e = tid + 256 * it;                  // coalesced f16 writes
        const int d = e >> 6, k = e & 63;
        VT[((size_t)bh * D_KEY + d) * S_LEN + kt * 64 + k] = tile[k][d];
    }
}

// ---------------- main attention kernel ----------------
__global__ __launch_bounds__(256, 1)
void attn_fwd_kernel(const float* __restrict__ Q,
                     const _Float16* __restrict__ K16,   // [bh][k][d] f16
                     const _Float16* __restrict__ VT,    // [bh][d][k] f16
                     const int*   __restrict__ mask,
                     float* __restrict__ out,            // [BH, S, D]
                     float* __restrict__ attn)           // [BH, S, S]
{
    extern __shared__ char smem[];
    _Float16* sh    = (_Float16*)smem;
    float*    shacc = (float*)(smem + NQ*S_LEN*2);
    float*    shinv = (float*)(smem + NQ*S_LEN*2 + NQ*D_KEY*4);

    const int tid  = threadIdx.x;
    const int wave = tid >> 5;
    const int lane = tid & 31;
    const int l16  = lane & 15;
    const int lh   = lane >> 4;          // half-wave select (WMMA operand layout)

    const int wg = blockIdx.x;
    const int qt = wg & (KT - 1);
    const int bh = wg >> 8;

    const float*    Qb  = Q   + (size_t)bh * S_LEN * D_KEY;
    const _Float16* Kb  = K16 + (size_t)bh * S_LEN * D_KEY;
    const _Float16* VTb = VT  + (size_t)bh * D_KEY * S_LEN;

    // ---- Q A-fragments (pre-scaled by 1/sqrt(D)=0.125), live in registers ----
    const float* qrow = Qb + (size_t)(qt * NQ + l16) * D_KEY;
    v16h a0, a1;
    {
        const int c0 = lh * 16;
        #pragma unroll
        for (int i = 0; i < 8; ++i) {
            a0[i]     = (_Float16)(qrow[c0 + i]           * 0.125f);
            a0[i + 8] = (_Float16)(qrow[c0 + 8 + i]       * 0.125f);
            a1[i]     = (_Float16)(qrow[32 + c0 + i]      * 0.125f);
            a1[i + 8] = (_Float16)(qrow[32 + c0 + 8 + i]  * 0.125f);
        }
    }

    // ---- GEMM1: scores = (Q*scale) @ K^T -> LDS f16 ----
    // B fragment: lane col n=l16, K rows lh*16..+15 == 16 contiguous f16 (32B aligned load)
    for (int t = wave; t < KT; t += WAVES) {
        const _Float16* kr = Kb + (size_t)(t * 16 + l16) * D_KEY + lh * 16;
        const v16h b0 = *(const v16h*)(kr);
        const v16h b1 = *(const v16h*)(kr + 32);
        v8f c = {};
        c = __builtin_amdgcn_wmma_f32_16x16x32_f16(false, a0, false, b0, (short)0, c, false, false);
        c = __builtin_amdgcn_wmma_f32_16x16x32_f16(false, a1, false, b1, (short)0, c, false, false);
        #pragma unroll
        for (int r = 0; r < 8; ++r) {
            const int m = r + lh * 8;
            sh[m * S_LEN + t * 16 + l16] = (_Float16)c[r];
        }
    }
    __syncthreads();

    // ---- Softmax: 16 threads per row, 4-wide vectors for the 1GB attention stream ----
    const int row = tid >> 4;
    const int j   = tid & 15;
    const int qg  = qt * NQ + row;
    const int* mrow = mask + (size_t)qg * S_LEN;
    _Float16* srow  = sh + row * S_LEN;

    float lmax = -3.0e38f;
    for (int t = 0; t < S_LEN / 64; ++t) {
        const int c0 = (t * 16 + j) * 4;
        const v4i mv = *(const v4i*)(mrow + c0);
        v4h s = *(const v4h*)(srow + c0);
        float f0 = (mv.x == 0) ? -60000.0f : (float)s[0];
        float f1 = (mv.y == 0) ? -60000.0f : (float)s[1];
        float f2 = (mv.z == 0) ? -60000.0f : (float)s[2];
        float f3 = (mv.w == 0) ? -60000.0f : (float)s[3];
        s[0] = (_Float16)f0; s[1] = (_Float16)f1; s[2] = (_Float16)f2; s[3] = (_Float16)f3;
        *(v4h*)(srow + c0) = s;
        lmax = fmaxf(fmaxf(fmaxf(lmax, f0), fmaxf(f1, f2)), f3);
    }
    #pragma unroll
    for (int o = 8; o >= 1; o >>= 1)
        lmax = fmaxf(lmax, __shfl_xor(lmax, o, 16));

    float lsum = 0.0f;
    for (int t = 0; t < S_LEN / 64; ++t) {
        const int c0 = (t * 16 + j) * 4;
        v4h s = *(const v4h*)(srow + c0);
        float e0 = __expf((float)s[0] - lmax);
        float e1 = __expf((float)s[1] - lmax);
        float e2 = __expf((float)s[2] - lmax);
        float e3 = __expf((float)s[3] - lmax);
        lsum += (e0 + e1) + (e2 + e3);
        s[0] = (_Float16)e0; s[1] = (_Float16)e1; s[2] = (_Float16)e2; s[3] = (_Float16)e3;
        *(v4h*)(srow + c0) = s;                       // keep unnormalized exp for GEMM2
    }
    #pragma unroll
    for (int o = 8; o >= 1; o >>= 1)
        lsum += __shfl_xor(lsum, o, 16);

    const float rinv = 1.0f / lsum;
    if (j == 0) shinv[row] = rinv;

    // zero cross-wave accumulator (1024 floats)
    #pragma unroll
    for (int i = 0; i < 4; ++i) shacc[tid * 4 + i] = 0.0f;

    // stream normalized attention to global: b128 non-temporal stores
    float* arow = attn + ((size_t)bh * S_LEN + qg) * S_LEN;
    for (int t = 0; t < S_LEN / 64; ++t) {
        const int c0 = (t * 16 + j) * 4;
        const v4h s = *(const v4h*)(srow + c0);
        v4f p;
        p.x = (float)s[0] * rinv; p.y = (float)s[1] * rinv;
        p.z = (float)s[2] * rinv; p.w = (float)s[3] * rinv;
        __builtin_nontemporal_store(p, (v4f*)(arow + c0));
    }
    __syncthreads();

    // ---- GEMM2: out = exp(P) @ V, per-row 1/sum scaling at the end ----
    v8f acc[4] = { v8f{}, v8f{}, v8f{}, v8f{} };
    for (int i = 0; i < 16; ++i) {
        const int kb   = (wave * 16 + i) * 32;       // this wave owns keys [wave*512, +512)
        const int koff = kb + lh * 16;
        const v8h p0 = *(const v8h*)(sh + l16 * S_LEN + koff);        // 16B-aligned LDS loads
        const v8h p1 = *(const v8h*)(sh + l16 * S_LEN + koff + 8);
        const v16h a = __builtin_shufflevector(p0, p1,
                       0,1,2,3,4,5,6,7,8,9,10,11,12,13,14,15);
        const int kr = kb + lh * 16;
        #pragma unroll
        for (int jt = 0; jt < 4; ++jt) {
            const int d = jt * 16 + l16;
            // V^T row d, 16 contiguous f16 (32B aligned) -> whole B fragment in 2 loads
            const v16h b = *(const v16h*)(VTb + (size_t)d * S_LEN + kr);
            acc[jt] = __builtin_amdgcn_wmma_f32_16x16x32_f16(false, a, false, b,
                                                             (short)0, acc[jt], false, false);
        }
    }
    // cross-wave reduction (ds_add_f32)
    #pragma unroll
    for (int jt = 0; jt < 4; ++jt) {
        #pragma unroll
        for (int r = 0; r < 8; ++r)
            atomicAdd(&shacc[(r + lh * 8) * D_KEY + jt * 16 + l16], acc[jt][r]);
    }
    __syncthreads();

    // per-row scale + coalesced v4 output store
    {
        const int m  = tid >> 4;
        const int d0 = (tid & 15) * 4;
        const float sc = shinv[m];
        float* orow = out + ((size_t)bh * S_LEN + qt * NQ + m) * D_KEY + d0;
        v4f o;
        o.x = shacc[m * D_KEY + d0 + 0] * sc;
        o.y = shacc[m * D_KEY + d0 + 1] * sc;
        o.z = shacc[m * D_KEY + d0 + 2] * sc;
        o.w = shacc[m * D_KEY + d0 + 3] * sc;
        __builtin_nontemporal_store(o, (v4f*)orow);
    }
}

extern "C" void kernel_launch(void* const* d_in, const int* in_sizes, int n_in,
                              void* d_out, int out_size, void* d_ws, size_t ws_size,
                              hipStream_t stream) {
    (void)in_sizes; (void)n_in; (void)ws_size; (void)out_size;
    const float* Qp = (const float*)d_in[0];
    const float* Kp = (const float*)d_in[1];
    const float* Vp = (const float*)d_in[2];
    const int*   Mp = (const int*)d_in[3];

    float* out  = (float*)d_out;                          // [16, 4096, 64]
    float* attn = out + (size_t)BH * S_LEN * D_KEY;       // [16, 4096, 4096]

    _Float16* K16 = (_Float16*)d_ws;                      // 8 MB
    _Float16* VT  = K16 + (size_t)BH * S_LEN * D_KEY;     // 8 MB

    cvt_kernel<<<BH * 64, 256, 0, stream>>>(Kp, Vp, K16, VT);

    (void)hipFuncSetAttribute((const void*)attn_fwd_kernel,
                              hipFuncAttributeMaxDynamicSharedMemorySize, SH_BYTES);
    attn_fwd_kernel<<<BH * (S_LEN / NQ), 256, SH_BYTES, stream>>>(Qp, K16, VT, Mp, out, attn);
}
